// Resnet_multiclassifier_49374944035210
// MI455X (gfx1250) — matrix operand order repaired and checked by reading, not compiled, not run on hardware
//
#include <hip/hip_runtime.h>
#include <stdint.h>

typedef __attribute__((ext_vector_type(16))) _Float16 v16h;
typedef __attribute__((ext_vector_type(8)))  _Float16 v8h;
typedef __attribute__((ext_vector_type(8)))  float    v8f;
typedef __attribute__((ext_vector_type(4)))  int      v4i;

#define BN_EPS   1e-5f
#define BATCH    1024
#define KCH      128              // K-chunk staged in LDS per barrier round
#define LDSROW   (KCH + 8)        // +16B row pad vs LDS bank conflicts (272B rows)

// CDNA5 async global->LDS staging (ASYNCcnt path), if the toolchain exposes it.
#if defined(__gfx1250__) && defined(__HIP_DEVICE_COMPILE__) && \
    __has_builtin(__builtin_amdgcn_global_load_async_to_lds_b128) && \
    __has_builtin(__builtin_amdgcn_s_wait_asynccnt)
#define HAVE_ASYNC_LDS 1
typedef __attribute__((address_space(1))) v4i* gv4ip;  // global v4i*
typedef __attribute__((address_space(3))) v4i* lv4ip;  // LDS v4i*
#else
#define HAVE_ASYNC_LDS 0
#endif

// ---------------------------------------------------------------------------
// BN-fold: W_f16[cout][k], k = (ky*ks+kx)*Cinp + cin, zero-padded; bias f32.
// ---------------------------------------------------------------------------
struct FoldP {
  const float* w; const float* g; const float* be; const float* m; const float* v;
  _Float16* wf; float* bias;
  int Cout, Cin, Cinp, ks, Kpad;
};

__global__ void fold_conv_kernel(FoldP p) {
  int i = blockIdx.x * blockDim.x + threadIdx.x;
  int total = p.Cout * p.Kpad;
  if (i >= total) return;
  int co = i / p.Kpad;
  int kk = i - co * p.Kpad;
  float s = p.g[co] * rsqrtf(p.v[co] + BN_EPS);
  if (kk == 0) p.bias[co] = p.be[co] - p.m[co] * s;
  int off = kk / p.Cinp;
  int cin = kk - off * p.Cinp;
  int K2 = p.ks * p.ks;
  float wv = 0.f;
  if (off < K2 && cin < p.Cin)
    wv = p.w[(size_t)(co * p.Cin + cin) * K2 + off] * s;  // OIHW source
  p.wf[(size_t)co * p.Kpad + kk] = (_Float16)wv;
}

// ---------------------------------------------------------------------------
// Pack input: f32 NCHW [B,3,32,32] -> f16 NHWC padded to 8 channels (zeros).
// ---------------------------------------------------------------------------
__global__ void pack_input_kernel(const float* x, _Float16* out, int C, int H, int W, int Cp) {
  long i = (long)blockIdx.x * blockDim.x + threadIdx.x;
  long total = (long)BATCH * H * W * Cp;
  if (i >= total) return;
  int c = (int)(i % Cp);
  long r = i / Cp;
  int xw = (int)(r % W); r /= W;
  int y  = (int)(r % H);
  int b  = (int)(r / H);
  float v = (c < C) ? x[(((size_t)b * C + c) * H + y) * W + xw] : 0.f;
  out[i] = (_Float16)v;
}

// ---------------------------------------------------------------------------
// Implicit-GEMM conv: one wave = 16 pixels x (NB*16) couts. K loop in chunks
// of 128 staged into LDS (async-to-LDS when available); inner step 32 with
// v_wmma_f32_16x16x32_f16; each A gather feeds NB WMMAs (register blocking
// in N). Fused bias + optional residual + ReLU, f16 NHWC in/out.
// ---------------------------------------------------------------------------
struct ConvP {
  const _Float16* in;
  const _Float16* wf;
  const float*    bias;
  const _Float16* res;
  _Float16*       out;
  long Mtiles;
  int Hin, Win, Hout, Wout, Cinp, Cout, K, Kpad;
  int stride, pad, ks;
  int cshift, wshift, hwshift;
  int relu, addres;
};

__device__ __forceinline__ v8h conv_load_a8(const ConvP& p, int b, int oy, int ox, int kk) {
  v8h z = {};
  if (kk >= p.K) return z;                       // K-tail padding
  int off = kk >> p.cshift;                      // Cinp is power of two
  int cin = kk & (p.Cinp - 1);
  int ky = 0, kx = 0;
  if (p.ks == 3) { ky = off / 3; kx = off - ky * 3; }
  int iy = oy * p.stride + ky - p.pad;
  int ix = ox * p.stride + kx - p.pad;
  if ((unsigned)iy >= (unsigned)p.Hin || (unsigned)ix >= (unsigned)p.Win) return z;  // spatial pad
  const _Float16* src = p.in + ((((size_t)b * p.Hin + iy) * p.Win) + ix) * p.Cinp + cin;
  return *(const v8h*)src;                       // 16B aligned: Cinp%8==0, cin%8==0
}

template <int NB>
__global__ __launch_bounds__(128) void conv_wmma_kernel(ConvP p) {
  __shared__ _Float16 ldsw[16 * NB][LDSROW];
  const int tidx = threadIdx.x;
  const int wave = tidx >> 5;
  const int lane = tidx & 31;
  const int half = lane >> 4;
  const int lid  = lane & 15;
  const int ntg  = blockIdx.y;                   // group of NB cout tiles

  long mt = (long)blockIdx.x * 4 + wave;         // wave-uniform
  const bool active = (mt < p.Mtiles);

  // A rows: lanes 0-15 and 16-31 both cover M=0..15 (different K halves).
  long m  = mt * 16 + lid;
  int b   = (int)(m >> p.hwshift);
  int rem = (int)(m & ((1 << p.hwshift) - 1));
  int oy  = rem >> p.wshift;
  int ox  = rem & ((1 << p.wshift) - 1);

  const _Float16* wbase = p.wf + (size_t)ntg * (16 * NB) * p.Kpad;
  const int tr = tidx >> 4;                      // staging: 8 rows in parallel,
  const int tk = (tidx & 15) * 8;                // 16 x v8h chunks per row

  v8f acc[NB];
#pragma unroll
  for (int j = 0; j < NB; ++j) acc[j] = (v8f){};

  for (int kc = 0; kc < p.Kpad; kc += KCH) {
    int kw = p.Kpad - kc; if (kw > KCH) kw = KCH;   // multiple of 32
    __syncthreads();                                // previous chunk fully read
    for (int r = tr; r < 16 * NB; r += 8)
      if (tk < kw) {
        const _Float16* g = wbase + (size_t)r * p.Kpad + kc + tk;
        _Float16* l = &ldsw[r][tk];
#if HAVE_ASYNC_LDS
        // cache -> LDS direct, no VGPR round-trip; tracked by ASYNCcnt
        __builtin_amdgcn_global_load_async_to_lds_b128(
            (gv4ip)(void*)const_cast<_Float16*>(g), (lv4ip)(void*)l, 0, 0);
#else
        *(v8h*)l = *(const v8h*)g;
#endif
      }
#if HAVE_ASYNC_LDS
    __builtin_amdgcn_s_wait_asynccnt(0);
#endif
    __syncthreads();

    if (active) {
      for (int k0 = 0; k0 < kw; k0 += 32) {
        // A layout: lanes 0-15: K = k0..k0+7 then k0+16..k0+23; lanes 16-31: +8.
        v8h a0 = conv_load_a8(p, b, oy, ox, kc + k0 + half * 8);
        v8h a1 = conv_load_a8(p, b, oy, ox, kc + k0 + 16 + half * 8);
        v16h af;
#pragma unroll
        for (int i = 0; i < 8; ++i) { af[i] = a0[i]; af[8 + i] = a1[i]; }
#pragma unroll
        for (int j = 0; j < NB; ++j) {
          // B: lane holds column N=lid of tile j, K = k0+half*8..+7 then +16..+23 (matches A).
          const _Float16* lrow = &ldsw[j * 16 + lid][k0 + half * 8];
          v8h b0 = *(const v8h*)lrow;
          v8h b1 = *(const v8h*)(lrow + 16);
          v16h bf;
#pragma unroll
          for (int i = 0; i < 8; ++i) { bf[i] = b0[i]; bf[8 + i] = b1[i]; }
          acc[j] = __builtin_amdgcn_wmma_f32_16x16x32_f16(false, af, false, bf,
                                                          (short)0, acc[j], false, false);
        }
      }
    }
  }

  if (!active) return;

  // C/D layout: lane<16 -> (M=r, N=lid); lane>=16 -> (M=r+8, N=lid).
#pragma unroll
  for (int j = 0; j < NB; ++j) {
    int n = (ntg * NB + j) * 16 + lid;
    float bsv = p.bias[n];
#pragma unroll
    for (int r = 0; r < 8; ++r) {
      long row = mt * 16 + half * 8 + r;
      int bb = (int)(row >> p.hwshift);
      int rr = (int)(row & ((1 << p.hwshift) - 1));
      int y  = rr >> p.wshift;
      int x2 = rr & ((1 << p.wshift) - 1);
      size_t oidx = (((size_t)bb << p.hwshift) + ((size_t)y << p.wshift) + x2) * p.Cout + n;
      float v = acc[j][r] + bsv;
      if (p.addres) v += (float)p.res[oidx];
      if (p.relu && v < 0.f) v = 0.f;
      p.out[oidx] = (_Float16)v;
    }
  }
}

// ---------------------------------------------------------------------------
// Global mean pool over 4x4 -> f16 feats [B,128]
// ---------------------------------------------------------------------------
__global__ void avgpool_kernel(const _Float16* in, _Float16* feats, int HW, int C) {
  int i = blockIdx.x * blockDim.x + threadIdx.x;
  if (i >= BATCH * C) return;
  int c = i % C, b = i / C;
  float s = 0.f;
  for (int q = 0; q < HW; ++q) s += (float)in[((size_t)b * HW + q) * C + c];
  feats[i] = (_Float16)(s * (1.f / HW));
}

// ---------------------------------------------------------------------------
// Task-gathered head: out[b,c] = W[tid[b]*C + c] . feats[b] + bias  (per-row
// distinct weights -> batched GEMV, not a shared GEMM; VALU is the right tool)
// ---------------------------------------------------------------------------
__global__ void fc_gather_kernel(const _Float16* feats, const float* W, const float* bias,
                                 const int* tid, float* out, int C, int D) {
  int i = blockIdx.x * blockDim.x + threadIdx.x;
  if (i >= BATCH * C) return;
  int c = i % C, b = i / C;
  int t = tid[b];
  const float* wr = W + (size_t)(t * C + c) * D;
  const _Float16* fr = feats + (size_t)b * D;
  float s = bias[t * C + c];
#pragma unroll 8
  for (int d = 0; d < D; ++d) s += wr[d] * (float)fr[d];
  out[i] = s;
}

// ---------------------------------------------------------------------------
// Host orchestration
// ---------------------------------------------------------------------------
static inline int ilog2i(int v) { int r = 0; while ((1 << r) < v) ++r; return r; }

extern "C" void kernel_launch(void* const* d_in, const int* in_sizes, int n_in,
                              void* d_out, int out_size, void* d_ws, size_t ws_size,
                              hipStream_t stream) {
  (void)in_sizes; (void)n_in; (void)out_size; (void)ws_size;

  // ---- walk d_in in setup_inputs() insertion order ----
  int ii = 0;
  const float* x   = (const float*)d_in[ii++];
  const int*   tid = (const int*)d_in[ii++];

  struct BN { const float *g, *b, *m, *v; };
  auto nextf  = [&]() { return (const float*)d_in[ii++]; };
  auto nextBN = [&]() { BN r; r.g = nextf(); r.b = nextf(); r.m = nextf(); r.v = nextf(); return r; };

  const float* conv1_w = nextf();
  BN bn1 = nextBN();

  struct Blk { int cin, cout, stride, has_sc;
               const float *c1w, *c2w, *scw; BN bn1, bn2, scbn; };
  Blk blk[8];
  const int cfg[4][3] = {{16,16,1},{16,32,2},{32,64,2},{64,128,2}};
  int nb = 0;
  for (int l = 0; l < 4; ++l) {
    for (int si = 0; si < 2; ++si) {
      Blk& B = blk[nb++];
      B.stride = (si == 0) ? cfg[l][2] : 1;
      B.cin    = (si == 0) ? cfg[l][0] : cfg[l][1];
      B.cout   = cfg[l][1];
      B.has_sc = (B.stride != 1 || B.cin != B.cout);
      B.c1w = nextf(); B.bn1 = nextBN();
      B.c2w = nextf(); B.bn2 = nextBN();
      B.scw = nullptr;
      if (B.has_sc) { B.scw = nextf(); B.scbn = nextBN(); }
    }
  }
  const float* fc_w = nextf();
  const float* fc_b = nextf();
  // remaining input: num_classes scalar (hardcoded 10)

  // ---- workspace carve-out (deterministic cursor, restarts every call) ----
  uintptr_t base = (uintptr_t)d_ws;
  size_t cur = 0;
  auto wsalloc = [&](size_t bytes) -> void* {
    cur = (cur + 255) & ~(size_t)255;
    void* p = (void*)(base + cur);
    cur += bytes;
    return p;
  };
  const size_t actBytes = (size_t)BATCH * 32 * 32 * 16 * sizeof(_Float16);  // 32 MiB
  _Float16* bufA = (_Float16*)wsalloc(actBytes);
  _Float16* bufB = (_Float16*)wsalloc(actBytes);
  _Float16* bufC = (_Float16*)wsalloc(actBytes);
  _Float16* feats = (_Float16*)wsalloc((size_t)BATCH * 128 * sizeof(_Float16));

  auto fold = [&](const float* w, BN bn, int Cout, int Cin, int Cinp, int ks,
                  const _Float16*& wf, const float*& bias) {
    int K = ks * ks * Cinp;
    int Kpad = (K + 31) & ~31;
    _Float16* wfp = (_Float16*)wsalloc((size_t)Cout * Kpad * sizeof(_Float16));
    float*    bp  = (float*)wsalloc((size_t)Cout * sizeof(float));
    FoldP fp; fp.w = w; fp.g = bn.g; fp.be = bn.b; fp.m = bn.m; fp.v = bn.v;
    fp.wf = wfp; fp.bias = bp; fp.Cout = Cout; fp.Cin = Cin; fp.Cinp = Cinp;
    fp.ks = ks; fp.Kpad = Kpad;
    int total = Cout * Kpad;
    fold_conv_kernel<<<dim3((unsigned)((total + 255) / 256)), 256, 0, stream>>>(fp);
    wf = wfp; bias = bp;
  };

  auto conv = [&](const _Float16* in, const _Float16* wf, const float* bias,
                  const _Float16* res, _Float16* out,
                  int Hin, int Cinp, int Cout, int ks, int stride, int relu) {
    ConvP p;
    p.in = in; p.wf = wf; p.bias = bias; p.res = res; p.out = out;
    p.Hin = Hin; p.Win = Hin;
    p.Hout = Hin / stride; p.Wout = p.Hout;
    p.Cinp = Cinp; p.Cout = Cout;
    p.K = ks * ks * Cinp; p.Kpad = (p.K + 31) & ~31;
    p.stride = stride; p.pad = (ks == 3) ? 1 : 0; p.ks = ks;
    p.cshift = ilog2i(Cinp);
    p.wshift = ilog2i(p.Wout);
    p.hwshift = ilog2i(p.Hout * p.Wout);
    p.relu = relu; p.addres = (res != nullptr) ? 1 : 0;
    long M = (long)BATCH * p.Hout * p.Wout;      // always a multiple of 16
    p.Mtiles = M >> 4;
    int NB = (Cout >= 64) ? 4 : (Cout == 32 ? 2 : 1);    // N register blocking
    int ntN = Cout / (16 * NB);
    long gx = (p.Mtiles + 3) / 4;                // 4 waves (one 16-row m-tile each)
    dim3 grid((unsigned)gx, (unsigned)ntN);
    if (NB == 4)      conv_wmma_kernel<4><<<grid, 128, 0, stream>>>(p);
    else if (NB == 2) conv_wmma_kernel<2><<<grid, 128, 0, stream>>>(p);
    else              conv_wmma_kernel<1><<<grid, 128, 0, stream>>>(p);
  };

  // ---- stem ----
  {
    long total = (long)BATCH * 32 * 32 * 8;
    pack_input_kernel<<<dim3((unsigned)((total + 255) / 256)), 256, 0, stream>>>(
        x, bufA, 3, 32, 32, 8);
  }
  const _Float16* wf; const float* bs;
  fold(conv1_w, bn1, 16, 3, 8, 3, wf, bs);
  conv(bufA, wf, bs, nullptr, bufB, 32, 8, 16, 3, 1, 1);

  // ---- residual blocks ----
  _Float16* curb = bufB;
  _Float16* f0 = bufA;
  _Float16* f1 = bufC;
  int H = 32;
  for (int i = 0; i < 8; ++i) {
    Blk& B = blk[i];
    const _Float16 *w1, *w2, *wsc = nullptr;
    const float *b1, *b2, *bsc = nullptr;
    fold(B.c1w, B.bn1, B.cout, B.cin, B.cin, 3, w1, b1);
    fold(B.c2w, B.bn2, B.cout, B.cout, B.cout, 3, w2, b2);
    if (B.has_sc) fold(B.scw, B.scbn, B.cout, B.cin, B.cin, 1, wsc, bsc);
    int Hout = H / B.stride;
    if (B.has_sc) {
      conv(curb, wsc, bsc, nullptr, f0, H, B.cin, B.cout, 1, B.stride, 0);  // shortcut
      conv(curb, w1,  b1,  nullptr, f1, H, B.cin, B.cout, 3, B.stride, 1);  // conv1+relu
      conv(f1,   w2,  b2,  f0,    curb, Hout, B.cout, B.cout, 3, 1, 1);     // conv2+sc+relu
      // curb stays; f0/f1 free again
    } else {
      conv(curb, w1, b1, nullptr, f0, H, B.cin,  B.cout, 3, 1, 1);          // conv1+relu
      conv(f0,   w2, b2, curb,    f1, H, B.cout, B.cout, 3, 1, 1);          // conv2+id+relu
      _Float16* t = curb; curb = f1; f1 = t;                                // rotate
    }
    H = Hout;
  }

  // ---- head ----
  avgpool_kernel<<<dim3((BATCH * 128 + 255) / 256), 256, 0, stream>>>(curb, feats, 16, 128);
  fc_gather_kernel<<<dim3((BATCH * 10 + 255) / 256), 256, 0, stream>>>(
      feats, fc_w, fc_b, tid, (float*)d_out, 10, 128);
}
